// Willmore_Adaptation_12644383719513
// MI455X (gfx1250) — matrix operand order repaired and checked by reading, not compiled, not run on hardware
//
#include <hip/hip_runtime.h>

// CDNA5 / gfx1250: depthwise causal 499-tap FIR via V_WMMA_F32_16X16X32_F16.
// D(16 out-times x 16 batches) += A(kernel-Toeplitz 16x32, persistent in LDS)
//                               * B(input 32x16 from LDS fp16 ring),
// accumulated over 17 tap chunks in the WMMA C operand.

typedef __attribute__((ext_vector_type(16))) _Float16 v16h;
typedef __attribute__((ext_vector_type(8)))  _Float16 v8h;
typedef __attribute__((ext_vector_type(8)))  float    v8f;

#define T_LEN   8000
#define F_LEN   64
#define KTOT    499
#define TQ      16      // output times per tile
#define NM      17      // 17 x 32 = 544 taps covers the 514-wide band
#define RING    1024    // fp16 ring depth per batch (power of two)
#define NB      16      // batches per workgroup (N dimension of WMMA)

__device__ __forceinline__ v8h cvt8(float4 lo, float4 hi) {
    v8h r;
    r[0] = (_Float16)lo.x; r[1] = (_Float16)lo.y;
    r[2] = (_Float16)lo.z; r[3] = (_Float16)lo.w;
    r[4] = (_Float16)hi.x; r[5] = (_Float16)hi.y;
    r[6] = (_Float16)hi.z; r[7] = (_Float16)hi.w;
    return r;
}

__global__ __launch_bounds__(32)
void fir_wmma_kernel(const float* __restrict__ x,
                     const float* __restrict__ a,
                     const float* __restrict__ w,
                     float* __restrict__ y,
                     int chunkLen)
{
    // A tiles built in exact A-operand lane order: [m][lane][16 halves]
    __shared__ _Float16 ldsA[NM * 32 * 16];
    __shared__ _Float16 ldsX[NB][RING];

    const int lane  = threadIdx.x;            // 0..31 (wave32)
    const int f     = blockIdx.x;             // feature
    const int bbase = blockIdx.y * NB;        // batch base (0 or 16)
    const int c0    = blockIdx.z * chunkLen;  // chunk start time

    // ---------------- per-feature FIR coefficients -> A tiles -------------
    // kern[tau] = -C*w*a^(497-tau) for tau<498, kern[498]=1, C=(1-a)/(1-a^498)
    const float av  = a[f];
    const float wv  = w[f];
    const float l2a = __builtin_amdgcn_logf(av);                      // log2(a)
    const float Cf  = (1.0f - av)
                    / (1.0f - __builtin_amdgcn_exp2f(498.0f * l2a));  // 2^x

    const int rowA = lane & 15;                     // A row (output-time offset i)
    const int kbA  = (lane < 16) ? 0 : 8;           // K base per half-wave (ISA layout)
    for (int m = 0; m < NM; ++m) {
        v16h vals;
        #pragma unroll
        for (int h = 0; h < 16; ++h) {
            const int K   = 2 * kbA + h;                   // A contiguous K order (matches B)
            const int tau = 32 * m + K - 14 - rowA;        // tap index for A_m[i][K]
            float cf = 0.0f;
            if (tau >= 0 && tau < KTOT)
                cf = (tau == KTOT - 1)
                   ? 1.0f
                   : (-Cf * wv * __builtin_amdgcn_exp2f((float)(497 - tau) * l2a));
            vals[h] = (_Float16)cf;
        }
        *(v16h*)&ldsA[(m * 32 + lane) * 16] = vals;        // 32B contiguous per lane
    }

    // ---------------- prefill fp16 ring: s in [c0-512, c0+32) -------------
    const int jb   = lane & 15;                             // batch column j
    const int half = lane >> 4;                             // 0/1
    const float* xj = x + ((size_t)(bbase + jb) * F_LEN + f) * T_LEN;
    for (int sb = c0 - 512 + half * 8; sb < c0 + 32; sb += 16) {
        v8h pk;
        if (sb < 0) {                                       // whole group left of t=0:
            _Float16 e = (_Float16)xj[0];                   // edge pad = x[0]
            #pragma unroll
            for (int q = 0; q < 8; ++q) pk[q] = e;
        } else {                                            // prologue never passes T-8
            float4 lo = *(const float4*)(xj + sb);
            float4 hi = *(const float4*)(xj + sb + 4);
            pk = cvt8(lo, hi);
        }
        *(v8h*)&ldsX[jb][(sb + 512) & (RING - 1)] = pk;     // one ds_store_b128
    }

    // ---------------- main loop over 16-output tiles -----------------------
    // Staging is pipelined one tile ahead: group [T0+32,T0+48) sits in registers
    // during tile T0's WMMA chain and is committed to the ring at iteration end.
    const int tiles = chunkLen / TQ;
    const int kloB  = (lane < 16) ? 0 : 16;                 // B contiguous K order: K half

    int    sThis = c0 + 32 + half * 8;                      // staged group base
    float4 r0, r1;
    {
        int sb = sThis;  if (sb > T_LEN - 8) sb = T_LEN - 8;   // dont-care tail cap
        r0 = *(const float4*)(xj + sb);
        r1 = *(const float4*)(xj + sb + 4);
    }

    for (int tile = 0; tile < tiles; ++tile) {
        const int T0 = c0 + tile * TQ;

        // issue next group's global loads (hidden behind the WMMA chain)
        const int sNext = T0 + 48 + half * 8;
        int sb = sNext;  if (sb > T_LEN - 8) sb = T_LEN - 8;
        __builtin_prefetch(xj + ((sb + 16 <= T_LEN - 8) ? sb + 16 : T_LEN - 8), 0, 1);
        float4 n0 = *(const float4*)(xj + sb);
        float4 n1 = *(const float4*)(xj + sb + 4);

        // 17 chained WMMAs: full 499-tap conv for this 16x16 tile
        v8f acc = {};
        #pragma unroll
        for (int m = 0; m < NM; ++m) {
            v16h Am = *(const v16h*)&ldsA[(m * 32 + lane) * 16];
            const int s = T0 - 512 + 32 * m + kloB;         // 16 contiguous times
            v16h Bm = *(const v16h*)&ldsX[jb][(s + 512) & (RING - 1)];
            acc = __builtin_amdgcn_wmma_f32_16x16x32_f16(
                      false, Am, false, Bm, (short)0, acc, false, false);
        }

        // commit previously staged group to the ring (needed from next tile on)
        *(v8h*)&ldsX[jb][(sThis + 512) & (RING - 1)] = cvt8(r0, r1);
        sThis = sNext;  r0 = n0;  r1 = n1;

        // |y| and store: lane -> batch jb, 8 consecutive times
        float* yrow = y + ((size_t)(bbase + jb) * F_LEN + f) * T_LEN
                        + T0 + ((lane < 16) ? 0 : 8);
        float4 olo, ohi;
        olo.x = __builtin_fabsf(acc[0]); olo.y = __builtin_fabsf(acc[1]);
        olo.z = __builtin_fabsf(acc[2]); olo.w = __builtin_fabsf(acc[3]);
        ohi.x = __builtin_fabsf(acc[4]); ohi.y = __builtin_fabsf(acc[5]);
        ohi.z = __builtin_fabsf(acc[6]); ohi.w = __builtin_fabsf(acc[7]);
        *reinterpret_cast<float4*>(yrow)     = olo;
        *reinterpret_cast<float4*>(yrow + 4) = ohi;
    }
}

extern "C" void kernel_launch(void* const* d_in, const int* in_sizes, int n_in,
                              void* d_out, int out_size, void* d_ws, size_t ws_size,
                              hipStream_t stream) {
    (void)in_sizes; (void)n_in; (void)out_size; (void)d_ws; (void)ws_size;
    const float* x = (const float*)d_in[0];   // (32,1,64,8000) fp32
    const float* a = (const float*)d_in[1];   // (64,)
    const float* w = (const float*)d_in[2];   // (64,)
    float* y = (float*)d_out;                 // (32,1,64,8000) fp32

    const int chunkLen = 800;                 // 10 chunks x 800 = 8000, 50 tiles each
    dim3 grid(F_LEN, 2, T_LEN / chunkLen);    // 64 features x 2 batch-halves x 10 chunks
    dim3 block(32);                           // one wave32 per workgroup
    hipLaunchKernelGGL(fir_wmma_kernel, grid, block, 0, stream, x, a, w, y, chunkLen);
}
